// SingleConstFilteredNormalized_42262478192688
// MI455X (gfx1250) — compile-verified, matrix operand order checked by reading, not run on hardware
//
#include <hip/hip_runtime.h>

// ---------------------------------------------------------------------------
// y[i,j] = (x[i,j] != 0) ? f / safe(f * count_i) : 0,  count_i = #nonzeros row i
// safe(s) = (s != 0) ? s : 1
//
// Memory-bound: 512 MB min traffic -> ~22 us floor @ 23.3 TB/s.
// Row counts computed on the matrix path: v_wmma_f32_16x16x32_f16 with a
// {0,1} f16 mask A-fragment and an all-ones B-fragment (f32 accumulate, exact
// for counts <= 4096). Pass 2 re-reads the 256 KB band from L2.
// ---------------------------------------------------------------------------

typedef __attribute__((ext_vector_type(16))) _Float16 v16h;
typedef __attribute__((ext_vector_type(8)))  float    v8f;

#define N_COLS          4096
#define ROWS_PER_BLOCK  16
#define THREADS         256
#define WAVES           (THREADS / 32)            // 8
#define COLS_PER_WAVE   (N_COLS / WAVES)          // 512
#define CHUNKS          (COLS_PER_WAVE / 32)      // 16 chunks of K=32

static __device__ __forceinline__ _Float16 nzh(float v) {
    return (v != 0.0f) ? (_Float16)1.0f : (_Float16)0.0f;
}

__global__ __launch_bounds__(THREADS)
void masked_norm_wmma_kernel(const float* __restrict__ x,
                             const float* __restrict__ fptr,
                             float* __restrict__ y) {
    __shared__ float cnt[ROWS_PER_BLOCK];

    const int tid  = threadIdx.x;
    const int lane = tid & 31;
    const int wave = tid >> 5;
    const int row  = lane & 15;      // M index within the 16-row band
    const int half = lane >> 4;      // which 16-col half of the K=32 chunk

    const long long grow = (long long)blockIdx.x * ROWS_PER_BLOCK + row;
    const float* __restrict__ rowp  = x + grow * (long long)N_COLS;
    float*       __restrict__ orowp = y + grow * (long long)N_COLS;

    // this lane's column start inside its wave's 512-col window
    const int cb0 = wave * COLS_PER_WAVE + half * 16;

    if (tid < ROWS_PER_BLOCK) cnt[tid] = 0.0f;
    __syncthreads();

    // B fragment: all ones (32x16 of f16) -> D[m][n] = row count, any n
    v16h bfrag;
#pragma unroll
    for (int i = 0; i < 16; ++i) bfrag[i] = (_Float16)1.0f;

    v8f acc = {};   // C/D accumulator (8 VGPRs, f32)

    // -------- pass 1: WMMA row-count over this wave's 512 columns ----------
#pragma unroll 4
    for (int c = 0; c < CHUNKS; ++c) {
        const float4* p = (const float4*)(rowp + cb0 + c * 32);
        float4 v0 = p[0], v1 = p[1], v2 = p[2], v3 = p[3];  // 4x b128

        v16h a;
        a[0]  = nzh(v0.x); a[1]  = nzh(v0.y); a[2]  = nzh(v0.z); a[3]  = nzh(v0.w);
        a[4]  = nzh(v1.x); a[5]  = nzh(v1.y); a[6]  = nzh(v1.z); a[7]  = nzh(v1.w);
        a[8]  = nzh(v2.x); a[9]  = nzh(v2.y); a[10] = nzh(v2.z); a[11] = nzh(v2.w);
        a[12] = nzh(v3.x); a[13] = nzh(v3.y); a[14] = nzh(v3.z); a[15] = nzh(v3.w);

        // D = A(16x32 f16 mask) x B(ones) + C   (K placement irrelevant: sum)
        acc = __builtin_amdgcn_wmma_f32_16x16x32_f16(
            /*neg_a=*/false, a, /*neg_b=*/false, bfrag,
            /*c_mod=*/(short)0, acc, /*reuse_a=*/false, /*reuse_b=*/false);
    }

    // D layout: VGPR r, lanes 0-15 -> M=r ; lanes 16-31 -> M=r+8 (all N equal)
    if (lane == 0) {
#pragma unroll
        for (int r = 0; r < 8; ++r) atomicAdd(&cnt[r], acc[r]);
    } else if (lane == 16) {
#pragma unroll
        for (int r = 0; r < 8; ++r) atomicAdd(&cnt[r + 8], acc[r]);
    }
    __syncthreads();

    // -------- per-row scalar: val = f / safe(f * count) --------------------
    const float f    = fptr[0];
    const float rs   = f * cnt[row];
    const float safe = (rs != 0.0f) ? rs : 1.0f;
    const float val  = f / safe;

    // -------- pass 2: stream band from L2, write masked val ----------------
#pragma unroll 4
    for (int c = 0; c < CHUNKS; ++c) {
        const float4* p = (const float4*)(rowp  + cb0 + c * 32);
        float4*       q = (float4*)      (orowp + cb0 + c * 32);
#pragma unroll
        for (int j = 0; j < 4; ++j) {
            float4 v = p[j];
            float4 o;
            o.x = (v.x != 0.0f) ? val : 0.0f;
            o.y = (v.y != 0.0f) ? val : 0.0f;
            o.z = (v.z != 0.0f) ? val : 0.0f;
            o.w = (v.w != 0.0f) ? val : 0.0f;
            q[j] = o;
        }
    }
}

extern "C" void kernel_launch(void* const* d_in, const int* in_sizes, int n_in,
                              void* d_out, int out_size, void* d_ws, size_t ws_size,
                              hipStream_t stream) {
    // setup_inputs order: {"t": scalar f32 (unused), "x": [B,N] f32, "f": [1] f32}
    const float* x = (const float*)d_in[1];
    const float* f = (const float*)d_in[2];
    float*       y = (float*)d_out;

    const int B = in_sizes[1] / N_COLS;           // 16384
    dim3 grid(B / ROWS_PER_BLOCK);                // 1024 blocks
    masked_norm_wmma_kernel<<<grid, THREADS, 0, stream>>>(x, f, y);
}
